// model_lstm_52578989638167
// MI455X (gfx1250) — matrix operand order repaired and checked by reading, not compiled, not run on hardware
//
#include <hip/hip_runtime.h>
#include <hip/hip_bf16.h>
#include <hip/hip_fp16.h>

typedef __attribute__((ext_vector_type(16))) _Float16 v16h;
typedef __attribute__((ext_vector_type(8)))  _Float16 v8h;
typedef __attribute__((ext_vector_type(8)))  float    v8f;

#define B_BATCH 128
#define T_SEQ   1024

__device__ __forceinline__ float sigf(float x) { return 1.0f / (1.0f + __expf(-x)); }
__device__ __forceinline__ float tanh_fast(float x) {
  float e = __expf(-2.0f * fabsf(x));
  float r = (1.0f - e) / (1.0f + e);
  return copysignf(r, x);
}

// Build padded, concatenated f16 weights: Wcat[4*Hp][Kp+Hp] row r = gate g, unit j.
// Columns [0,Kp) = w_ih (padded from Din), [Kp,Kp+Hp) = w_hh (padded from H).
// bias[r] = b_ih + b_hh (0 in padding). Optional wihv = w_ih column 0 (layer-1 Din=1).
__global__ void build_wcat(const float* __restrict__ w_ih, const float* __restrict__ w_hh,
                           const float* __restrict__ b_ih, const float* __restrict__ b_hh,
                           int H, int Din, int Hp, int Kp,
                           _Float16* __restrict__ Wcat, float* __restrict__ bias,
                           float* __restrict__ wihv) {
  const int KTOT = Kp + Hp;
  int r = blockIdx.x;                 // [0, 4*Hp)
  int g = r / Hp, j = r - g * Hp;
  bool valid = (j < H);
  int srow = g * H + j;
  for (int k = threadIdx.x; k < KTOT; k += blockDim.x) {
    float v = 0.0f;
    if (valid) {
      if (k < Kp) { if (k < Din) v = w_ih[(size_t)srow * Din + k]; }
      else { int kh = k - Kp; if (kh < H) v = w_hh[(size_t)srow * H + kh]; }
    }
    Wcat[(size_t)r * KTOT + k] = (_Float16)v;
  }
  if (threadIdx.x == 0) {
    bias[r] = valid ? (b_ih[srow] + b_hh[srow]) : 0.0f;
    if (wihv) wihv[r] = valid ? w_ih[(size_t)srow * Din] : 0.0f;
  }
}

// Persistent LSTM scan. Each WG owns HS hidden units (all four gates for them).
// KP: padded input dim (0 for layer 1), HP: padded hidden dim. KTOT=KP+HP.
// 256 threads = 8 waves; wave w owns batch rows [16w,16w+16). Cell state c stays
// in registers in WMMA C-fragment layout for all T steps. h ping-pongs through L2;
// one device-wide monotonic counter barrier per step across the <=10 WGs.
template <int KP, int HP, int HS, bool XS>
__global__ __launch_bounds__(256) void lstm_scan(
    const _Float16* __restrict__ xin,   // [B][T][KP] f16   (XS == false)
    const float*    __restrict__ x1,    // [B][T] f32       (XS == true, Din==1)
    const _Float16* __restrict__ Wcat,  // [4*HP][KTOT] f16
    const float*    __restrict__ bias,  // [4*HP]
    const float*    __restrict__ wihv,  // [4*HP] (XS only)
    _Float16*       __restrict__ hbuf,  // [2][B][HP] ping-pong h state
    _Float16*       __restrict__ yout,  // [B][T][HP]
    int*            __restrict__ bar) {
  constexpr int KTOT = KP + HP;
  constexpr int NT4  = (4 * HS) / 16;   // gate-column tiles per WG
  constexpr int JT   = HS / 16;         // hidden-unit tiles per WG
  constexpr int B    = B_BATCH;
  constexpr int T    = T_SEQ;

  extern __shared__ char smem[];
  _Float16* Alds = (_Float16*)smem;                      // [B][KTOT]  (x_t || h_{t-1})
  _Float16* Wlds = Alds + (size_t)B * KTOT;              // [4*HS][KTOT]
  float*    xsh  = (float*)(Wlds + (size_t)4 * HS * KTOT);

  const int wg   = blockIdx.x;
  const int nwg  = gridDim.x;
  const int tid  = threadIdx.x;
  const int wave = tid >> 5;
  const int lane = tid & 31;
  const int n    = lane & 15;   // column within 16-wide N tile (C/D layout)
  const int hf   = lane >> 4;   // half-select (A: +8 rows in C, +16 K in A/B)

  // Stage this WG's weight slice into LDS once (resident for all 1024 steps).
  for (int idx = tid; idx < 4 * HS * (KTOT / 8); idx += 256) {
    int row = idx / (KTOT / 8);
    int c8  = idx - row * (KTOT / 8);
    int g = row / HS, s = row - g * HS;
    ((v8h*)(Wlds + (size_t)row * KTOT))[c8] =
        ((const v8h*)(Wcat + (size_t)(g * HP + wg * HS + s) * KTOT))[c8];
  }

  float breg[4][JT], wreg[4][JT];
#pragma unroll
  for (int g = 0; g < 4; ++g)
#pragma unroll
    for (int jt = 0; jt < JT; ++jt) {
      int jg = wg * HS + jt * 16 + n;
      breg[g][jt] = bias[g * HP + jg];
      wreg[g][jt] = XS ? wihv[g * HP + jg] : 0.0f;
    }

  float cst[JT][8];
#pragma unroll
  for (int jt = 0; jt < JT; ++jt)
#pragma unroll
    for (int r = 0; r < 8; ++r) cst[jt][r] = 0.0f;

  const v8f vzero = {};
  __syncthreads();

  for (int t = 0; t < T; ++t) {
    const _Float16* hin  = hbuf + (size_t)(t & 1) * B * HP;
    _Float16*       hout = hbuf + (size_t)((t + 1) & 1) * B * HP;

    // Stage A = [x_t | h_{t-1}] into LDS (reused by 8 waves x NT4 tiles).
    if constexpr (!XS) {
      for (int idx = tid; idx < B * (KP / 8); idx += 256) {
        int b = idx / (KP / 8), c8 = idx - b * (KP / 8);
        ((v8h*)(Alds + (size_t)b * KTOT))[c8] =
            ((const v8h*)(xin + ((size_t)b * T + t) * KP))[c8];
      }
      if (t + 1 < T && tid < B)
        __builtin_prefetch(xin + ((size_t)tid * T + (t + 1)) * KP, 0, 1);
    } else {
      if (tid < B) xsh[tid] = x1[(size_t)tid * T + t];
    }
    for (int idx = tid; idx < B * (HP / 8); idx += 256) {
      int b = idx / (HP / 8), c8 = idx - b * (HP / 8);
      ((v8h*)(Alds + (size_t)b * KTOT + KP))[c8] =
          ((const v8h*)(hin + (size_t)b * HP))[c8];
    }
    __syncthreads();

    v8f acc[NT4];
#pragma unroll
    for (int nt = 0; nt < NT4; ++nt) acc[nt] = vzero;

    // gates[16w..16w+16, wg-slice] = A . Wslice^T via v_wmma_f32_16x16x32_f16
    const _Float16* arow = Alds + (size_t)(wave * 16 + n) * KTOT;
    for (int kt = 0; kt < KTOT / 32; ++kt) {
      const int k0 = kt * 32;
      // A 16x32 f16, contiguous K order used here (B matches): lane<16 holds K in [0,16); lane>=16 holds K in [16,32).
      v8h a0 = *(const v8h*)(arow + k0 + hf * 16);
      v8h a1 = *(const v8h*)(arow + k0 + hf * 16 + 8);
      v16h af = __builtin_shufflevector(a0, a1, 0, 1, 2, 3, 4, 5, 6, 7,
                                        8, 9, 10, 11, 12, 13, 14, 15);
#pragma unroll
      for (int nt = 0; nt < NT4; ++nt) {
        // B 32x16: lane holds 16 contiguous K of column n -> contiguous row of W.
        v16h bf = *(const v16h*)(Wlds + (size_t)(nt * 16 + n) * KTOT + k0 + hf * 16);
        acc[nt] = __builtin_amdgcn_wmma_f32_16x16x32_f16(false, af, false, bf,
                                                         (short)0, acc[nt], false, false);
      }
    }

    float xv[8];
    if constexpr (XS) {
#pragma unroll
      for (int r = 0; r < 8; ++r) xv[r] = xsh[wave * 16 + hf * 8 + r];
    }

    // Fused epilogue: i/f/g/o for unit j are in the same lane across 4 fragments.
#pragma unroll
    for (int jt = 0; jt < JT; ++jt) {
      v8f gi = acc[0 * JT + jt];
      v8f gf = acc[1 * JT + jt];
      v8f gg = acc[2 * JT + jt];
      v8f go = acc[3 * JT + jt];
      const int jg = wg * HS + jt * 16 + n;
#pragma unroll
      for (int r = 0; r < 8; ++r) {
        float xi = XS ? xv[r] : 0.0f;
        float ig = sigf(gi[r] + breg[0][jt] + (XS ? xi * wreg[0][jt] : 0.0f));
        float fg = sigf(gf[r] + breg[1][jt] + (XS ? xi * wreg[1][jt] : 0.0f));
        float cg = tanh_fast(gg[r] + breg[2][jt] + (XS ? xi * wreg[2][jt] : 0.0f));
        float og = sigf(go[r] + breg[3][jt] + (XS ? xi * wreg[3][jt] : 0.0f));
        float c = fg * cst[jt][r] + ig * cg;
        float h = og * tanh_fast(c);
        cst[jt][r] = c;
        int mg = wave * 16 + hf * 8 + r;   // C/D layout: row = 8*(lane>=16)+vgpr
        _Float16 hq = (_Float16)h;
        hout[(size_t)mg * HP + jg] = hq;
        yout[((size_t)mg * T + t) * HP + jg] = hq;
      }
    }

    // Device-wide step barrier (monotonic counter; h is ping-ponged so one suffices).
    __threadfence();
    __syncthreads();
    if (tid == 0) {
      __atomic_fetch_add(bar, 1, __ATOMIC_RELEASE);
      const int target = nwg * (t + 1);
      while (__atomic_load_n(bar, __ATOMIC_ACQUIRE) < target)
        __builtin_amdgcn_s_sleep(1);
    }
    __syncthreads();
    __threadfence();
  }
}

// Layer 4: H=1 -> trivial VALU recurrence, one lane per batch row. Only final h needed.
__global__ __launch_bounds__(128) void lstm_l4(
    const _Float16* __restrict__ y3,   // [B][T][64] f16
    const float* __restrict__ w_ih,    // [4][64]
    const float* __restrict__ w_hh,    // [4][1]
    const float* __restrict__ b_ih, const float* __restrict__ b_hh,
    float* __restrict__ out) {
  __shared__ float wsl[4 * 64];
  int tid = threadIdx.x;
  for (int i = tid; i < 256; i += 128) wsl[i] = w_ih[i];
  __syncthreads();
  if (tid >= B_BATCH) return;
  float bi = b_ih[0] + b_hh[0], bf = b_ih[1] + b_hh[1];
  float bg = b_ih[2] + b_hh[2], bo = b_ih[3] + b_hh[3];
  float whi = w_hh[0], whf = w_hh[1], whg = w_hh[2], who = w_hh[3];
  float h = 0.0f, c = 0.0f;
  for (int t = 0; t < T_SEQ; ++t) {
    const _Float16* xr = y3 + ((size_t)tid * T_SEQ + t) * 64;
    float ai = 0.0f, af = 0.0f, ag = 0.0f, ao = 0.0f;
#pragma unroll 8
    for (int k = 0; k < 64; ++k) {
      float x = (float)xr[k];
      ai += x * wsl[k];
      af += x * wsl[64 + k];
      ag += x * wsl[128 + k];
      ao += x * wsl[192 + k];
    }
    float ig = sigf(ai + bi + whi * h);
    float fg = sigf(af + bf + whf * h);
    float cg = tanh_fast(ag + bg + whg * h);
    float og = sigf(ao + bo + who * h);
    c = fg * c + ig * cg;
    h = og * tanh_fast(c);
  }
  out[tid] = h;
}

extern "C" void kernel_launch(void* const* d_in, const int* in_sizes, int n_in,
                              void* d_out, int out_size, void* d_ws, size_t ws_size,
                              hipStream_t stream) {
  (void)in_sizes; (void)n_in; (void)out_size; (void)ws_size;
  const float* x    = (const float*)d_in[0];
  const float* wih1 = (const float*)d_in[1];
  const float* whh1 = (const float*)d_in[2];
  const float* bih1 = (const float*)d_in[3];
  const float* bhh1 = (const float*)d_in[4];
  const float* wih2 = (const float*)d_in[5];
  const float* whh2 = (const float*)d_in[6];
  const float* bih2 = (const float*)d_in[7];
  const float* bhh2 = (const float*)d_in[8];
  const float* wih3 = (const float*)d_in[9];
  const float* whh3 = (const float*)d_in[10];
  const float* bih3 = (const float*)d_in[11];
  const float* bhh3 = (const float*)d_in[12];
  const float* wih4 = (const float*)d_in[13];
  const float* whh4 = (const float*)d_in[14];
  const float* bih4 = (const float*)d_in[15];
  const float* bhh4 = (const float*)d_in[16];
  float* out = (float*)d_out;

  char* ws = (char*)d_ws;
  size_t off = 0;
  auto alloc = [&](size_t bytes) -> char* {
    char* p = ws + off;
    off = (off + bytes + 255) & ~(size_t)255;
    return p;
  };

  int*      bar1 = (int*)alloc(256);
  int*      bar2 = (int*)alloc(256);
  int*      bar3 = (int*)alloc(256);
  _Float16* h1   = (_Float16*)alloc((size_t)2 * 128 * 320 * 2);
  _Float16* h2   = (_Float16*)alloc((size_t)2 * 128 * 256 * 2);
  _Float16* h3   = (_Float16*)alloc((size_t)2 * 128 * 64 * 2);
  _Float16* Wc1  = (_Float16*)alloc((size_t)1280 * 320 * 2);
  float*    b1   = (float*)alloc(1280 * 4);
  float*    wv1  = (float*)alloc(1280 * 4);
  _Float16* Wc2  = (_Float16*)alloc((size_t)1024 * 576 * 2);
  float*    b2   = (float*)alloc(1024 * 4);
  _Float16* Wc3  = (_Float16*)alloc((size_t)256 * 320 * 2);
  float*    b3   = (float*)alloc(256 * 4);
  _Float16* y1   = (_Float16*)alloc((size_t)128 * 1024 * 320 * 2);
  _Float16* y2   = (_Float16*)alloc((size_t)128 * 1024 * 256 * 2);
  _Float16* y3   = (_Float16*)alloc((size_t)128 * 1024 * 64 * 2);

  hipMemsetAsync(bar1, 0, 256, stream);
  hipMemsetAsync(bar2, 0, 256, stream);
  hipMemsetAsync(bar3, 0, 256, stream);
  hipMemsetAsync(h1, 0, (size_t)2 * 128 * 320 * 2, stream);
  hipMemsetAsync(h2, 0, (size_t)2 * 128 * 256 * 2, stream);
  hipMemsetAsync(h3, 0, (size_t)2 * 128 * 64 * 2, stream);

  // Padded f16 weight concatenations + fused biases.
  build_wcat<<<dim3(4 * 320), dim3(256), 0, stream>>>(wih1, whh1, bih1, bhh1,
                                                      300, 1, 320, 0, Wc1, b1, wv1);
  build_wcat<<<dim3(4 * 256), dim3(256), 0, stream>>>(wih2, whh2, bih2, bhh2,
                                                      256, 300, 256, 320, Wc2, b2, nullptr);
  build_wcat<<<dim3(4 * 64), dim3(256), 0, stream>>>(wih3, whh3, bih3, bhh3,
                                                     64, 256, 64, 256, Wc3, b3, nullptr);

  (void)hipFuncSetAttribute((const void*)lstm_scan<0, 320, 32, true>,
                            hipFuncAttributeMaxDynamicSharedMemorySize, 320 * 1024);
  (void)hipFuncSetAttribute((const void*)lstm_scan<320, 256, 32, false>,
                            hipFuncAttributeMaxDynamicSharedMemorySize, 320 * 1024);
  (void)hipFuncSetAttribute((const void*)lstm_scan<256, 64, 32, false>,
                            hipFuncAttributeMaxDynamicSharedMemorySize, 320 * 1024);

  // Dynamic LDS: A[B][KTOT] + W[4*HS][KTOT] (f16) + xs[B] (f32)
  size_t s1 = ((size_t)128 * 320 + (size_t)128 * 320) * 2 + 512;   // 160.5 KB
  size_t s2 = ((size_t)128 * 576 + (size_t)128 * 576) * 2 + 512;   // 288.5 KB (<320 KB WGP LDS)
  size_t s3 = ((size_t)128 * 320 + (size_t)128 * 320) * 2 + 512;   // 160.5 KB

  lstm_scan<0, 320, 32, true><<<dim3(10), dim3(256), s1, stream>>>(
      nullptr, x, Wc1, b1, wv1, h1, y1, bar1);
  lstm_scan<320, 256, 32, false><<<dim3(8), dim3(256), s2, stream>>>(
      y1, nullptr, Wc2, b2, nullptr, h2, y2, bar2);
  lstm_scan<256, 64, 32, false><<<dim3(2), dim3(256), s3, stream>>>(
      y2, nullptr, Wc3, b3, nullptr, h3, y3, bar3);
  lstm_l4<<<dim3(1), dim3(128), 0, stream>>>(y3, wih4, whh4, bih4, bhh4, out);
}
